// Graph_Enhance_model_16106127360686
// MI455X (gfx1250) — matrix operand order repaired and checked by reading, not compiled, not run on hardware
//
#include <hip/hip_runtime.h>

// ---- problem constants (from reference) ----
#define D_DIM  2048
#define HD     1024      // D/2
#define G3     6144      // 3*D
#define F_DIM  32
#define H_DIM  8
#define O_DIM  16
#define M_EDGE 16384     // B*F*H*O
#define M_BFH  1024      // B*F*H
#define M_BF   128       // B*F
#define M_BFO  2048      // B*F*O

typedef __attribute__((ext_vector_type(16))) __bf16         v16bf;
typedef __attribute__((ext_vector_type(8)))  float          v8f;
typedef __attribute__((ext_vector_type(8)))  unsigned short v8us;
typedef __attribute__((ext_vector_type(16))) unsigned short v16us;

__device__ __forceinline__ unsigned short f2bf(float f) {
    unsigned u = __builtin_bit_cast(unsigned, f);
    u += 0x7FFFu + ((u >> 16) & 1u);          // round-to-nearest-even
    return (unsigned short)(u >> 16);
}
__device__ __forceinline__ v8us pack8(float4 x, float4 y) {
    v8us o;
    o[0]=f2bf(x.x); o[1]=f2bf(x.y); o[2]=f2bf(x.z); o[3]=f2bf(x.w);
    o[4]=f2bf(y.x); o[5]=f2bf(y.y); o[6]=f2bf(y.z); o[7]=f2bf(y.w);
    return o;
}
__device__ __forceinline__ v16bf frag16(v8us lo, v8us hi) {
    v16us u = __builtin_shufflevector(lo, hi, 0,1,2,3,4,5,6,7,8,9,10,11,12,13,14,15);
    return __builtin_bit_cast(v16bf, u);
}
__device__ __forceinline__ float sigm(float x) { return 1.f / (1.f + __expf(-x)); }

// ============================================================================
// WMMA GEMM: C[M,N] = A[M,K] @ W[N,K]^T + bias.  A, W are PRE-CONVERTED bf16.
// Block = 256 threads (8 waves), 128x128 tile; wave computes 32x64 (8 wmma).
// Double-buffered LDS: one barrier per K-step, global loads overlap WMMA.
// MODE 0: C[m,n]   = acc + bias                         (fp32 out, stride ldc)
// MODE 1: fused relu -> *wl2 -> 16-col reduce -> atomicAdd(logits[m])
// MODE 2: t=(acc+bias)*wrow[m]; Eout[m, n] = bf16(t)    (stride D_DIM, lo half)
// MODE 3: t=(acc+bias)*wrow[m]; Msum[m>>4, n] = bf16(mean over 16 rows == O)
// ============================================================================
template<int MODE>
__global__ __launch_bounds__(256)
void gemm_wmma(const unsigned short* __restrict__ A, int lda,
               const unsigned short* __restrict__ W, const float* __restrict__ bias,
               float* __restrict__ C, int ldc, int K,
               const float* __restrict__ wl2v, float* __restrict__ logits,
               const float* __restrict__ wrow,
               unsigned short* __restrict__ Eout, unsigned short* __restrict__ Msum)
{
    constexpr int LDSW = 40;                         // ushorts per row (pad)
    __shared__ __attribute__((aligned(16))) unsigned short ldsA[2][128 * LDSW];
    __shared__ __attribute__((aligned(16))) unsigned short ldsW[2][128 * LDSW];

    const int tid   = threadIdx.x;
    const int lane  = tid & 31;
    const int wave  = tid >> 5;
    const int m0    = blockIdx.y * 128;
    const int n0    = blockIdx.x * 128;
    const int mwave = wave >> 1;                     // 0..3 -> 32-row strip
    const int nwave = wave & 1;                      // 0..1 -> 64-col strip

    v8f acc[2][4] = {};

    // cooperative staging: thread -> row lr (0..127), 16-col half lc (0/16)
    const int lr = tid >> 1;
    const int lc = (tid & 1) << 4;
    const int lofs = lr * LDSW + lc;
    const unsigned short* Ag = A + (size_t)(m0 + lr) * lda + lc;
    const unsigned short* Wg = W + (size_t)(n0 + lr) * (size_t)K + lc;

    // fragment addressing (ISA 16-bit A/B VGPR layouts, wave32)
    const int klo = (lane >> 4) << 3;                // A frag: K 0..7 / 8..15 (+16)
    const int kb  = (lane >> 4) << 4;                // unused: B frag uses klo, same K order as A
    const int arow0 = (mwave * 32 + (lane & 15)) * LDSW;
    const int brow0 = (nwave * 64 + (lane & 15)) * LDSW;

    // prologue: stage tile k0=0 into buffer 0
    v8us a0 = *(const v8us*)(Ag);
    v8us a1 = *(const v8us*)(Ag + 8);
    v8us w0 = *(const v8us*)(Wg);
    v8us w1 = *(const v8us*)(Wg + 8);
    *(v8us*)&ldsA[0][lofs]     = a0;  *(v8us*)&ldsA[0][lofs + 8] = a1;
    *(v8us*)&ldsW[0][lofs]     = w0;  *(v8us*)&ldsW[0][lofs + 8] = w1;

    int p = 0;
    for (int k0 = 0; k0 < K; k0 += 32) {
        __syncthreads();
        const bool more = (k0 + 32) < K;
        if (more) {                                  // fetch next tile (overlaps wmma)
            a0 = *(const v8us*)(Ag + k0 + 32);
            a1 = *(const v8us*)(Ag + k0 + 40);
            w0 = *(const v8us*)(Wg + k0 + 32);
            w1 = *(const v8us*)(Wg + k0 + 40);
        }
        v16bf af[2], bf[4];
        #pragma unroll
        for (int mi = 0; mi < 2; ++mi) {
            const int ro = arow0 + mi * 16 * LDSW;
            af[mi] = frag16(*(const v8us*)&ldsA[p][ro + klo],
                            *(const v8us*)&ldsA[p][ro + 16 + klo]);
        }
        #pragma unroll
        for (int ni = 0; ni < 4; ++ni) {
            const int ro = brow0 + ni * 16 * LDSW;
            bf[ni] = frag16(*(const v8us*)&ldsW[p][ro + klo],
                            *(const v8us*)&ldsW[p][ro + 16 + klo]);
        }
        #pragma unroll
        for (int mi = 0; mi < 2; ++mi)
            #pragma unroll
            for (int ni = 0; ni < 4; ++ni)
                acc[mi][ni] = __builtin_amdgcn_wmma_f32_16x16x32_bf16(
                    false, af[mi], false, bf[ni], (short)0, acc[mi][ni], false, false);
        if (more) {
            *(v8us*)&ldsA[p ^ 1][lofs]     = a0;  *(v8us*)&ldsA[p ^ 1][lofs + 8] = a1;
            *(v8us*)&ldsW[p ^ 1][lofs]     = w0;  *(v8us*)&ldsW[p ^ 1][lofs + 8] = w1;
        }
        p ^= 1;
    }

    // ---- epilogue (C layout: lane holds col=lane&15; VGPR r -> row r+8*(lane>>4)) ----
    const int col   = lane & 15;
    const int rhalf = lane >> 4;
    #pragma unroll
    for (int mi = 0; mi < 2; ++mi) {
        const int gm0    = m0 + mwave * 32 + mi * 16;    // 16-aligned == one (b,f,h)
        const int gmBase = gm0 + (rhalf << 3);
        #pragma unroll
        for (int ni = 0; ni < 4; ++ni) {
            v8f a = acc[mi][ni];
            const int gn = n0 + nwave * 64 + ni * 16 + col;
            const float bv = bias[gn];
            if (MODE == 0) {
                #pragma unroll
                for (int r = 0; r < 8; ++r)
                    C[(size_t)(gmBase + r) * ldc + gn] = a[r] + bv;
            } else if (MODE == 1) {
                const float wl = wl2v[gn];
                #pragma unroll
                for (int r = 0; r < 8; ++r) {
                    float v = a[r] + bv;
                    v = v > 0.f ? v * wl : 0.f;
                    v += __shfl_xor(v, 1, 32);
                    v += __shfl_xor(v, 2, 32);
                    v += __shfl_xor(v, 4, 32);
                    v += __shfl_xor(v, 8, 32);
                    if (col == 0) atomicAdd(&logits[gmBase + r], v);
                }
            } else {
                float s = 0.f;
                #pragma unroll
                for (int r = 0; r < 8; ++r) {
                    const int gm = gmBase + r;
                    const float t = (a[r] + bv) * wrow[gm];
                    if (MODE == 2) Eout[(size_t)gm * D_DIM + gn] = f2bf(t);
                    s += t;
                }
                if (MODE == 3) {
                    s += __shfl_xor(s, 16, 32);
                    if (rhalf == 0) Msum[(size_t)(gm0 >> 4) * D_DIM + gn] = f2bf(s * 0.0625f);
                }
            }
        }
    }
}

// ============================================================================
// Small kernels
// ============================================================================
__global__ void cvt_bf16(const float* __restrict__ s, unsigned short* __restrict__ d, int n8) {
    int i = blockIdx.x * blockDim.x + threadIdx.x;
    if (i >= n8) return;
    const float4* q = (const float4*)(s + (size_t)i * 8);
    *(v8us*)(d + (size_t)i * 8) = pack8(q[0], q[1]);
}

__global__ void fill_from(float* __restrict__ p, const float* __restrict__ v, int n) {
    int i = blockIdx.x * blockDim.x + threadIdx.x;
    if (i < n) p[i] = v[0];
}

__global__ void softmax16(const float* __restrict__ logits, float* __restrict__ w, int groups) {
    int g = blockIdx.x * blockDim.x + threadIdx.x;
    if (g >= groups) return;
    const float* L = logits + g * 16;
    float mx = L[0];
    #pragma unroll
    for (int o = 1; o < 16; ++o) mx = fmaxf(mx, L[o]);
    float e[16]; float s = 0.f;
    #pragma unroll
    for (int o = 0; o < 16; ++o) { e[o] = __expf(L[o] - mx); s += e[o]; }
    const float inv = 1.f / s;
    #pragma unroll
    for (int o = 0; o < 16; ++o) w[g * 16 + o] = e[o] * inv;
}

// E1 high half: bf16( wrow1[m] * msgn[bf*O+o, j] )   (E_hi = w1 (x) msg_n)
__global__ void e1_hi_fill(const float* __restrict__ wrow, const float* __restrict__ msgn,
                           unsigned short* __restrict__ E1) {
    int idx = blockIdx.x * blockDim.x + threadIdx.x;       // M_EDGE * HD
    if (idx >= M_EDGE * HD) return;
    int gm = idx >> 10, j = idx & (HD - 1);
    int o = gm & 15, bfi = gm >> 7;
    E1[(size_t)gm * D_DIM + HD + j] = f2bf(wrow[gm] * msgn[(size_t)((bfi << 4) + o) * HD + j]);
}

// Msum high half: bf16( (1/16) * sum_o w2[bfh,o] * msgn[bf*O+o, j] )
__global__ void msum_hi_kernel(const float* __restrict__ wrow, const float* __restrict__ msgn,
                               unsigned short* __restrict__ Msum) {
    int idx = blockIdx.x * blockDim.x + threadIdx.x;       // M_BFH * HD
    if (idx >= M_BFH * HD) return;
    int bfh = idx >> 10, j = idx & (HD - 1);
    int bf = bfh >> 3;
    float s = 0.f;
    #pragma unroll
    for (int o = 0; o < O_DIM; ++o)
        s += wrow[bfh * 16 + o] * msgn[(size_t)((bf << 4) + o) * HD + j];
    Msum[(size_t)bfh * D_DIM + HD + j] = f2bf(s * 0.0625f);
}

// gates ordered [r, z, n] per PyTorch GRU; fp32 and/or bf16 outputs
__global__ void gru_combine(const float* __restrict__ gi, const float* __restrict__ gh,
                            const float* __restrict__ h, float* __restrict__ out,
                            unsigned short* __restrict__ bfout, int total) {
    int idx = blockIdx.x * blockDim.x + threadIdx.x;
    if (idx >= total) return;
    int row = idx >> 11, d = idx & (D_DIM - 1);
    size_t b3 = (size_t)row * G3;
    float r = sigm(gi[b3 + d]              + gh[b3 + d]);
    float z = sigm(gi[b3 + D_DIM + d]      + gh[b3 + D_DIM + d]);
    float n = tanhf(gi[b3 + 2 * D_DIM + d] + r * gh[b3 + 2 * D_DIM + d]);
    float hp = h[(size_t)row * D_DIM + d];
    float v = (1.f - z) * n + z * hp;
    if (out)   out[(size_t)row * D_DIM + d]   = v;
    if (bfout) bfout[(size_t)row * D_DIM + d] = f2bf(v);
}

__global__ void mean_h_kernel(const float* __restrict__ humans, unsigned short* __restrict__ allh) {
    int idx = blockIdx.x * blockDim.x + threadIdx.x;       // M_BF * D
    if (idx >= M_BF * D_DIM) return;
    int bf = idx >> 11, d = idx & (D_DIM - 1);
    float s = 0.f;
    #pragma unroll
    for (int h = 0; h < H_DIM; ++h) s += humans[(size_t)((bf << 3) + h) * D_DIM + d];
    allh[idx] = f2bf(s * 0.125f);
}

__global__ void transpose_sf(const float* __restrict__ fs, float* __restrict__ sf,
                             unsigned short* __restrict__ sfb) {
    int idx = blockIdx.x * blockDim.x + threadIdx.x;       // B*F*D
    if (idx >= M_BF * D_DIM) return;
    int b = idx / (F_DIM * D_DIM);
    int rem = idx - b * (F_DIM * D_DIM);
    int f = rem >> 11, d = rem & (D_DIM - 1);
    float v = fs[(size_t)b * D_DIM * F_DIM + (size_t)d * F_DIM + f];
    sf[idx]  = v;
    sfb[idx] = f2bf(v);
}

// ============================================================================
extern "C" void kernel_launch(void* const* d_in, const int* in_sizes, int n_in,
                              void* d_out, int out_size, void* d_ws, size_t ws_size,
                              hipStream_t stream)
{
    (void)in_sizes; (void)n_in; (void)out_size; (void)ws_size;
    const float* S_node  = (const float*)d_in[0];
    const float* finalS  = (const float*)d_in[1];
    const float* H_nodes = (const float*)d_in[2];
    const float* O_nodes = (const float*)d_in[3];
    const float* E0      = (const float*)d_in[4];
    const float* Wn  = (const float*)d_in[5];  const float* bn  = (const float*)d_in[6];
    const float* We  = (const float*)d_in[7];  const float* be  = (const float*)d_in[8];
    const float* Wl1 = (const float*)d_in[9];  const float* bl1 = (const float*)d_in[10];
    const float* Wl2 = (const float*)d_in[11]; const float* bl2 = (const float*)d_in[12];
    const float* gh_wih = (const float*)d_in[13]; const float* gh_whh = (const float*)d_in[14];
    const float* gh_bih = (const float*)d_in[15]; const float* gh_bhh = (const float*)d_in[16];
    const float* gs_wih = (const float*)d_in[17]; const float* gs_whh = (const float*)d_in[18];
    const float* gs_bih = (const float*)d_in[19]; const float* gs_bhh = (const float*)d_in[20];
    float* out = (float*)d_out;

    // ---- fp32 scratch ----
    float* wf    = (float*)d_ws;
    float* msgn  = wf; wf += (size_t)M_BFO * HD;      // O_nodes @ Wn^T + bn (fp32: reused in fp32 math)
    float* ghh   = wf; wf += (size_t)M_BFH * G3;      // H_nodes @ gh_whh^T + bhh
    float* gih   = wf; wf += (size_t)M_BFH * G3;
    float* humans= wf; wf += (size_t)M_BFH * D_DIM;
    float* logits= wf; wf += M_EDGE;
    float* wrow1 = wf; wf += M_EDGE;
    float* wrow2 = wf; wf += M_EDGE;
    float* Sf    = wf; wf += (size_t)M_BF * D_DIM;
    // ---- bf16 scratch (16B-aligned continuation) ----
    unsigned short* us = (unsigned short*)wf;
    unsigned short* bWn    = us; us += (size_t)HD * D_DIM;
    unsigned short* bWe    = us; us += (size_t)HD * D_DIM;
    unsigned short* bWl1   = us; us += (size_t)HD * D_DIM;
    unsigned short* bGhWhh = us; us += (size_t)G3 * D_DIM;
    unsigned short* bGhWih = us; us += (size_t)G3 * D_DIM;
    unsigned short* bGsWih = us; us += (size_t)G3 * D_DIM;
    unsigned short* bGsWhh = us; us += (size_t)G3 * D_DIM;
    unsigned short* bE0    = us; us += (size_t)M_EDGE * D_DIM;
    unsigned short* bE1    = us; us += (size_t)M_EDGE * D_DIM;
    unsigned short* bMsum  = us; us += (size_t)M_BFH * D_DIM;
    unsigned short* bHnod  = us; us += (size_t)M_BFH * D_DIM;
    unsigned short* bOnod  = us; us += (size_t)M_BFO * D_DIM;
    unsigned short* bAllh  = us; us += (size_t)M_BF * D_DIM;
    unsigned short* bSnode = us; us += (size_t)M_BF * D_DIM;
    unsigned short* bS1    = us; us += (size_t)M_BF * D_DIM;
    unsigned short* bSf    = us; us += (size_t)M_BF * D_DIM;

    const dim3 blk(256);
    #define CVT(src, dst, n) cvt_bf16<<<((n)/8 + 255)/256, blk, 0, stream>>>(src, dst, (n)/8)

    // ---- one-time bf16 conversion of all GEMM operands ----
    CVT(Wn,      bWn,    HD * D_DIM);
    CVT(We,      bWe,    HD * D_DIM);
    CVT(Wl1,     bWl1,   HD * D_DIM);
    CVT(gh_whh,  bGhWhh, G3 * D_DIM);
    CVT(gh_wih,  bGhWih, G3 * D_DIM);
    CVT(gs_wih,  bGsWih, G3 * D_DIM);
    CVT(gs_whh,  bGsWhh, G3 * D_DIM);
    CVT(E0,      bE0,    M_EDGE * D_DIM);
    CVT(H_nodes, bHnod,  M_BFH * D_DIM);
    CVT(O_nodes, bOnod,  M_BFO * D_DIM);
    CVT(S_node,  bSnode, M_BF * D_DIM);
    transpose_sf<<<(M_BF * D_DIM + 255) / 256, blk, 0, stream>>>(finalS, Sf, bSf);

    // ---- loop-invariant precompute ----
    gemm_wmma<0><<<dim3(HD/128, M_BFO/128), blk, 0, stream>>>(
        bOnod, D_DIM, bWn, bn, msgn, HD, D_DIM, nullptr, nullptr, nullptr, nullptr, nullptr);
    gemm_wmma<0><<<dim3(G3/128, M_BFH/128), blk, 0, stream>>>(
        bHnod, D_DIM, bGhWhh, gh_bhh, ghh, G3, D_DIM, nullptr, nullptr, nullptr, nullptr, nullptr);

    // ---- iteration 1: only the new edges matter (humans_1 is dead) ----
    fill_from<<<(M_EDGE + 255) / 256, blk, 0, stream>>>(logits, bl2, M_EDGE);
    gemm_wmma<1><<<dim3(HD/128, M_EDGE/128), blk, 0, stream>>>(
        bE0, D_DIM, bWl1, bl1, nullptr, 0, D_DIM, Wl2, logits, nullptr, nullptr, nullptr);
    softmax16<<<(M_BFH + 255) / 256, blk, 0, stream>>>(logits, wrow1, M_BFH);
    gemm_wmma<2><<<dim3(HD/128, M_EDGE/128), blk, 0, stream>>>(
        bE0, D_DIM, bWe, be, nullptr, 0, D_DIM, nullptr, nullptr, wrow1, bE1, nullptr);
    e1_hi_fill<<<(M_EDGE * HD + 255) / 256, blk, 0, stream>>>(wrow1, msgn, bE1);

    // ---- iteration 2: only Msum is needed (edges die after this) ----
    fill_from<<<(M_EDGE + 255) / 256, blk, 0, stream>>>(logits, bl2, M_EDGE);
    gemm_wmma<1><<<dim3(HD/128, M_EDGE/128), blk, 0, stream>>>(
        bE1, D_DIM, bWl1, bl1, nullptr, 0, D_DIM, Wl2, logits, nullptr, nullptr, nullptr);
    softmax16<<<(M_BFH + 255) / 256, blk, 0, stream>>>(logits, wrow2, M_BFH);
    gemm_wmma<3><<<dim3(HD/128, M_EDGE/128), blk, 0, stream>>>(
        bE1, D_DIM, bWe, be, nullptr, 0, D_DIM, nullptr, nullptr, wrow2, nullptr, bMsum);
    msum_hi_kernel<<<(M_BFH * HD + 255) / 256, blk, 0, stream>>>(wrow2, msgn, bMsum);

    gemm_wmma<0><<<dim3(G3/128, M_BFH/128), blk, 0, stream>>>(
        bMsum, D_DIM, bGhWih, gh_bih, gih, G3, D_DIM, nullptr, nullptr, nullptr, nullptr, nullptr);
    gru_combine<<<(M_BFH * D_DIM + 255) / 256, blk, 0, stream>>>(
        gih, ghh, H_nodes, humans, nullptr, M_BFH * D_DIM);

    // ---- readout: two stacked S-GRUs (reuse gih/ghh head as gate scratch) ----
    float* gis = gih;                  // 128 x 6144 fits inside gih
    float* ghs = ghh;                  // 128 x 6144 fits inside ghh
    mean_h_kernel<<<(M_BF * D_DIM + 255) / 256, blk, 0, stream>>>(humans, bAllh);
    gemm_wmma<0><<<dim3(G3/128, 1), blk, 0, stream>>>(
        bAllh, D_DIM, bGsWih, gs_bih, gis, G3, D_DIM, nullptr, nullptr, nullptr, nullptr, nullptr);
    gemm_wmma<0><<<dim3(G3/128, 1), blk, 0, stream>>>(
        bSnode, D_DIM, bGsWhh, gs_bhh, ghs, G3, D_DIM, nullptr, nullptr, nullptr, nullptr, nullptr);
    gru_combine<<<(M_BF * D_DIM + 255) / 256, blk, 0, stream>>>(
        gis, ghs, S_node, nullptr, bS1, M_BF * D_DIM);
    gemm_wmma<0><<<dim3(G3/128, 1), blk, 0, stream>>>(
        bS1, D_DIM, bGsWih, gs_bih, gis, G3, D_DIM, nullptr, nullptr, nullptr, nullptr, nullptr);
    gemm_wmma<0><<<dim3(G3/128, 1), blk, 0, stream>>>(
        bSf, D_DIM, bGsWhh, gs_bhh, ghs, G3, D_DIM, nullptr, nullptr, nullptr, nullptr, nullptr);
    gru_combine<<<(M_BF * D_DIM + 255) / 256, blk, 0, stream>>>(
        gis, ghs, Sf, out, nullptr, M_BF * D_DIM);
    #undef CVT
}
